// CAM_58291296141385
// MI455X (gfx1250) — matrix operand order repaired and checked by reading, not compiled, not run on hardware
//
#include <hip/hip_runtime.h>

typedef _Float16 half_t;
typedef half_t v16h __attribute__((ext_vector_type(16)));
typedef float  v8f  __attribute__((ext_vector_type(8)));
typedef unsigned int uint32;
typedef uint32 v4u __attribute__((ext_vector_type(4)));
typedef uint32 v2u __attribute__((ext_vector_type(2)));

#define ROWS_PER_BLK 64          // 4 waves x 16 rows
#define RSH 424                  // staging row stride in halfs (=212 dwords)

// ---- LDS layout (dword offsets) ----
// Weight arrays are quad-packed: v2u element [q][n] holds f16 W[4q..4q+3][n]
enum {
  OFF_E1  = 0,                   // 24 x 132 v2u (K=96  pad from 88,  N=128)
  OFF_E2  = 6336,                // 48 x 132 v2u (K=192 pad from 168, N=128)
  OFF_AFA = 19008,               // 64 x 132 v2u (K=256, N=128)
  OFF_AFV = 35904,               // 64 x 132 v2u
  OFF_WA  = 52800,               // 64 x 36  v2u (K=256=[att;plain], N=32)
  OFF_WV  = 57408,               // 64 x 36  v2u
  OFF_HA  = 62016,               // 40 x 20  v2u (K=160=[h;base], N=16 pad from 8)
  OFF_HV  = 63616,               // 40 x 20  v2u
  OFF_R1  = 65216,               // 8 x 132  v2u (K=32 pad from 16, N=128)
  OFF_E1B = 67328,               // f32 biases / vectors
  OFF_E2B = 67456,
  OFF_E3B = 67584,
  OFF_E4B = 67600,
  OFF_R1B = 67616,
  OFF_R2W = 67744,
  OFF_NORM= 67872,               // 4 waves x 32 f32 (inv norms)
  OFF_STG = 68000,               // 4 waves x 16 x 212
  LDS_DWORDS = 81568
};
#define LDS_BYTES (LDS_DWORDS * 4)   // 326272 B  (<= 320KB WGP LDS)

__device__ __forceinline__ uint32 pk(float a, float b) {
  union { half_t h[2]; uint32 u; } x;
  x.h[0] = (half_t)a; x.h[1] = (half_t)b; return x.u;
}
__device__ __forceinline__ v2u pkq(float a, float b, float c, float d) {
  v2u r; r.x = pk(a, b); r.y = pk(c, d); return r;
}
__device__ __forceinline__ void st4h(half_t* p, float4 v) {
  union { half_t h[4]; unsigned long long u; } x;
  x.h[0] = (half_t)v.x; x.h[1] = (half_t)v.y;
  x.h[2] = (half_t)v.z; x.h[3] = (half_t)v.w;
  *(unsigned long long*)p = x.u;
}
__device__ __forceinline__ void st4z(half_t* p) { *(unsigned long long*)p = 0ull; }

// compile-time + DS-counter fence for cross-lane LDS reuse boundaries
__device__ __forceinline__ void lds_fence() {
  __asm__ volatile("s_wait_dscnt 0x0" ::: "memory");
}

// CDNA5 has native TRANS tanh; fall back to exp2-based rational (no Inf/Inf)
__device__ __forceinline__ float fast_tanh(float x) {
#if __has_builtin(__builtin_amdgcn_tanhf)
  return __builtin_amdgcn_tanhf(x);
#else
  float cx = fminf(fmaxf(x, -9.f), 9.f);
  float e = __builtin_amdgcn_exp2f(cx * 2.8853900817779268f);  // exp(2x)
  return (e - 1.f) * __builtin_amdgcn_rcpf(e + 1.f);
#endif
}

__device__ __forceinline__ v8f wmma32(v16h a, v16h b, v8f c) {
  return __builtin_amdgcn_wmma_f32_16x16x32_f16(false, a, false, b, (short)0, c,
                                                false, false);
}

// A fragment (16x32 f16): lane L<16 holds row L, K {0..7,16..23}; lanes>=16: +8
__device__ __forceinline__ v16h load_a(const half_t* stg, int colofs, int lane) {
  const half_t* p = stg + (lane & 15) * RSH + colofs + ((lane >> 4) << 3);
  union { v16h v; v4u q[2]; } a;
  a.q[0] = *(const v4u*)p;
  a.q[1] = *(const v4u*)(p + 16);
  return a.v;
}
// B fragment (32x16 f16) from quad-packed array Wq[q][n]: 4 x ds_load_b64,
// each b64 yields fragment dwords (u[2t],u[2t+1]) already in order -> no moves
__device__ __forceinline__ v16h load_b(const v2u* W, int stride, int qbase,
                                       int nb, int lane) {
  const v2u* p = W + (qbase + ((lane >> 4) << 1)) * stride + nb * 16 + (lane & 15);
  union { v16h v; v2u d[4]; } b;
#pragma unroll
  for (int j = 0; j < 4; ++j) b.d[j] = p[(j + (j & 2)) * stride];
  return b.v;
}

__global__ __launch_bounds__(128)
void cam_fused(const float* __restrict__ f1, const float* __restrict__ f2,
               const float* __restrict__ e1_w, const float* __restrict__ e1_b,
               const float* __restrict__ e2_w, const float* __restrict__ e2_b,
               const float* __restrict__ aff_a, const float* __restrict__ aff_v,
               const float* __restrict__ w_a,  const float* __restrict__ w_v,
               const float* __restrict__ w_ca, const float* __restrict__ w_cv,
               const float* __restrict__ w_ha, const float* __restrict__ w_hv,
               const float* __restrict__ e3_w, const float* __restrict__ e3_b,
               const float* __restrict__ e4_w, const float* __restrict__ e4_b,
               const float* __restrict__ r1_w, const float* __restrict__ r1_b,
               const float* __restrict__ r2_w, const float* __restrict__ r2_b,
               float* __restrict__ out, int Btot, int ntiles) {
  extern __shared__ __align__(16) uint32 lds[];
  float* fl = (float*)lds;
  const int tid = threadIdx.x;

  // ============ prolog: stage all weights as quad-packed f16 B-fragments ====
  {
    v2u* W = (v2u*)(lds + OFF_E1);                    // e1_w (128,88) -> K96
    for (int i = tid; i < 24 * 128; i += 128) {
      int q = i >> 7, n = i & 127, k = 4 * q;
      W[q * 132 + n] = pkq(k < 88 ? e1_w[n * 88 + k] : 0.f,
                           k + 1 < 88 ? e1_w[n * 88 + k + 1] : 0.f,
                           k + 2 < 88 ? e1_w[n * 88 + k + 2] : 0.f,
                           k + 3 < 88 ? e1_w[n * 88 + k + 3] : 0.f);
    }
  }
  {
    v2u* W = (v2u*)(lds + OFF_E2);                    // e2_w (128,168) -> K192
    for (int i = tid; i < 48 * 128; i += 128) {
      int q = i >> 7, n = i & 127, k = 4 * q;
      W[q * 132 + n] = pkq(k < 168 ? e2_w[n * 168 + k] : 0.f,
                           k + 1 < 168 ? e2_w[n * 168 + k + 1] : 0.f,
                           k + 2 < 168 ? e2_w[n * 168 + k + 2] : 0.f,
                           k + 3 < 168 ? e2_w[n * 168 + k + 3] : 0.f);
    }
  }
  {
    v2u* Wa = (v2u*)(lds + OFF_AFA);                  // aff_a/aff_v (128,256)
    v2u* Wv = (v2u*)(lds + OFF_AFV);
    for (int i = tid; i < 64 * 128; i += 128) {
      int q = i >> 7, n = i & 127, k = 4 * q;
      const float* pa = aff_a + n * 256 + k;
      const float* pv = aff_v + n * 256 + k;
      Wa[q * 132 + n] = pkq(pa[0], pa[1], pa[2], pa[3]);
      Wv[q * 132 + n] = pkq(pv[0], pv[1], pv[2], pv[3]);
    }
  }
  {
    v2u* Wa = (v2u*)(lds + OFF_WA);                   // [w_c*;w_*] (32,256 cat)
    v2u* Wv = (v2u*)(lds + OFF_WV);
    for (int i = tid; i < 64 * 32; i += 128) {
      int q = i >> 5, n = i & 31, k = 4 * q;          // quads never straddle 128
      const float* pa; const float* pv;
      if (k < 128) { pa = w_ca + n * 128 + k;       pv = w_cv + n * 128 + k; }
      else         { pa = w_a  + n * 128 + (k-128); pv = w_v  + n * 128 + (k-128); }
      Wa[q * 36 + n] = pkq(pa[0], pa[1], pa[2], pa[3]);
      Wv[q * 36 + n] = pkq(pv[0], pv[1], pv[2], pv[3]);
    }
  }
  {
    v2u* Wa = (v2u*)(lds + OFF_HA);                   // [w_h*;e3/4_w] (8,160 cat)
    v2u* Wv = (v2u*)(lds + OFF_HV);
    for (int i = tid; i < 40 * 16; i += 128) {
      int q = i >> 4, n = i & 15, k = 4 * q;          // quads never straddle 32
      v2u a = pkq(0.f, 0.f, 0.f, 0.f), v = a;
      if (n < 8) {
        const float* pa; const float* pv;
        if (k < 32) { pa = w_ha + n * 32 + k;      pv = w_hv + n * 32 + k; }
        else        { pa = e3_w + n * 128 + (k-32); pv = e4_w + n * 128 + (k-32); }
        a = pkq(pa[0], pa[1], pa[2], pa[3]);
        v = pkq(pv[0], pv[1], pv[2], pv[3]);
      }
      Wa[q * 20 + n] = a;
      Wv[q * 20 + n] = v;
    }
  }
  {
    v2u* W = (v2u*)(lds + OFF_R1);                    // r1_w (128,16) -> K32
    for (int i = tid; i < 8 * 128; i += 128) {
      int q = i >> 7, n = i & 127, k = 4 * q;         // valid only k<16
      W[q * 132 + n] = pkq(k < 16 ? r1_w[n * 16 + k] : 0.f,
                           k + 1 < 16 ? r1_w[n * 16 + k + 1] : 0.f,
                           k + 2 < 16 ? r1_w[n * 16 + k + 2] : 0.f,
                           k + 3 < 16 ? r1_w[n * 16 + k + 3] : 0.f);
    }
  }
  for (int i = tid; i < 128; i += 128) {
    fl[OFF_E1B + i] = e1_b[i]; fl[OFF_E2B + i] = e2_b[i];
    fl[OFF_R1B + i] = r1_b[i]; fl[OFF_R2W + i] = r2_w[i];
  }
  for (int i = tid; i < 16; i += 128) {
    fl[OFF_E3B + i] = (i < 8) ? e3_b[i] : 0.f;
    fl[OFF_E4B + i] = (i < 8) ? e4_b[i] : 0.f;
  }
  __syncthreads();

  // ================= persistent tile loop ====================================
  const int wave = tid >> 5, lane = tid & 31;
  const int n16 = lane & 15, hi = lane >> 4;
  half_t* stg = (half_t*)(lds + OFF_STG + wave * (16 * (RSH / 2)));
  float* NRM = fl + OFF_NORM + wave * 32;
  const v2u* E1q  = (const v2u*)(lds + OFF_E1);
  const v2u* E2q  = (const v2u*)(lds + OFF_E2);
  const v2u* AFAq = (const v2u*)(lds + OFF_AFA);
  const v2u* AFVq = (const v2u*)(lds + OFF_AFV);
  const v2u* WAq  = (const v2u*)(lds + OFF_WA);
  const v2u* WVq  = (const v2u*)(lds + OFF_WV);
  const v2u* HAq  = (const v2u*)(lds + OFF_HA);
  const v2u* HVq  = (const v2u*)(lds + OFF_HV);
  const v2u* R1q  = (const v2u*)(lds + OFF_R1);
  const float r2bv = r2_b[0];

  for (int tile = blockIdx.x; tile < ntiles; tile += gridDim.x) {
    const int row0 = tile * ROWS_PER_BLK + wave * 16;
    const int r = lane >> 1, pp = lane & 1;
    long row = row0 + r; if (row > Btot - 1) row = Btot - 1;

    // ---- stage f1/f2 raw as f16, accumulate squared norms (2 lanes/row) ----
    {
      const float4* src = (const float4*)(f1 + (size_t)row * 88);
      float ss = 0.f;
#pragma unroll
      for (int t = 0; t < 11; ++t) {
        float4 v = src[2 * t + pp];
        ss = fmaf(v.x, v.x, fmaf(v.y, v.y, fmaf(v.z, v.z, fmaf(v.w, v.w, ss))));
        st4h(stg + r * RSH + 8 * t + 4 * pp, v);
      }
      st4z(stg + r * RSH + 88 + 4 * pp);                 // pad K 88..95
      ss += __shfl_xor(ss, 1, 32);
      if (pp == 0) NRM[r] = 1.f / fmaxf(sqrtf(ss), 1e-12f);
    }
    {
      const float4* src = (const float4*)(f2 + (size_t)row * 168);
      float ss = 0.f;
#pragma unroll
      for (int t = 0; t < 21; ++t) {
        float4 v = src[2 * t + pp];
        ss = fmaf(v.x, v.x, fmaf(v.y, v.y, fmaf(v.z, v.z, fmaf(v.w, v.w, ss))));
        st4h(stg + r * RSH + 96 + 8 * t + 4 * pp, v);
      }
      st4z(stg + r * RSH + 264 + 12 * pp);               // pad K 168..191
      st4z(stg + r * RSH + 268 + 12 * pp);
      st4z(stg + r * RSH + 272 + 12 * pp);
      ss += __shfl_xor(ss, 1, 32);
      if (pp == 0) NRM[16 + r] = 1.f / fmaxf(sqrtf(ss), 1e-12f);
    }
    lds_fence();

    // ---- aud = f1n @ e1^T + b ; vis = f2n @ e2^T + b (norm applied post-GEMM)
    v8f zero = {};
    v8f cA[8], cV[8];
#pragma unroll
    for (int nb = 0; nb < 8; ++nb) { cA[nb] = zero; cV[nb] = zero; }
#pragma unroll
    for (int kb = 0; kb < 3; ++kb) {
      v16h a = load_a(stg, kb * 32, lane);
#pragma unroll
      for (int nb = 0; nb < 8; ++nb)
        cA[nb] = wmma32(a, load_b(E1q, 132, kb * 8, nb, lane), cA[nb]);
    }
#pragma unroll
    for (int kb = 0; kb < 6; ++kb) {
      v16h a = load_a(stg, 96 + kb * 32, lane);
#pragma unroll
      for (int nb = 0; nb < 8; ++nb)
        cV[nb] = wmma32(a, load_b(E2q, 132, kb * 8, nb, lane), cV[nb]);
    }
    // epilogue: scale by inv-norm, add bias, stage av=[aud,vis] as f16
#pragma unroll
    for (int g = 0; g < 8; ++g) {
      const int M = g + hi * 8;
      const float i1 = NRM[M], i2 = NRM[16 + M];
#pragma unroll
      for (int nb = 0; nb < 8; ++nb) {
        const int N = nb * 16 + n16;
        float aud = cA[nb][g] * i1 + fl[OFF_E1B + N];
        float vis = cV[nb][g] * i2 + fl[OFF_E2B + N];
        cA[nb][g] = aud; cV[nb][g] = vis;
        stg[M * RSH + N] = (half_t)aud;
        stg[M * RSH + 128 + N] = (half_t)vis;
      }
    }
    lds_fence();

    // ---- a_t = av @ aff_a^T ; audio_att = tanh(aud*a_t/16) ; h_a -----------
    v8f cT[8];
#pragma unroll
    for (int nb = 0; nb < 8; ++nb) cT[nb] = zero;
#pragma unroll
    for (int kb = 0; kb < 8; ++kb) {
      v16h a = load_a(stg, kb * 32, lane);
#pragma unroll
      for (int nb = 0; nb < 8; ++nb)
        cT[nb] = wmma32(a, load_b(AFAq, 132, kb * 8, nb, lane), cT[nb]);
    }
#pragma unroll
    for (int g = 0; g < 8; ++g)
#pragma unroll
      for (int nb = 0; nb < 8; ++nb)
        stg[(g + hi * 8) * RSH + 288 + nb * 16 + n16] =
            (half_t)fast_tanh(cA[nb][g] * cT[nb][g] * 0.0625f);
    lds_fence();

    v8f cHa0 = zero, cHa1 = zero;
#pragma unroll
    for (int kb = 0; kb < 8; ++kb) {
      v16h a = load_a(stg, kb < 4 ? 288 + kb * 32 : (kb - 4) * 32, lane);
      cHa0 = wmma32(a, load_b(WAq, 36, kb * 8, 0, lane), cHa0);
      cHa1 = wmma32(a, load_b(WAq, 36, kb * 8, 1, lane), cHa1);
    }

    // ---- v_t = av @ aff_v^T ; vis_att ; h_v --------------------------------
#pragma unroll
    for (int nb = 0; nb < 8; ++nb) cT[nb] = zero;
#pragma unroll
    for (int kb = 0; kb < 8; ++kb) {
      v16h a = load_a(stg, kb * 32, lane);
#pragma unroll
      for (int nb = 0; nb < 8; ++nb)
        cT[nb] = wmma32(a, load_b(AFVq, 132, kb * 8, nb, lane), cT[nb]);
    }
    lds_fence();                                    // h_a reads of 288 done
#pragma unroll
    for (int g = 0; g < 8; ++g)
#pragma unroll
      for (int nb = 0; nb < 8; ++nb)
        stg[(g + hi * 8) * RSH + 288 + nb * 16 + n16] =
            (half_t)fast_tanh(cV[nb][g] * cT[nb][g] * 0.0625f);
    lds_fence();

    v8f cHv0 = zero, cHv1 = zero;
#pragma unroll
    for (int kb = 0; kb < 8; ++kb) {
      v16h a = load_a(stg, kb < 4 ? 288 + kb * 32 : 128 + (kb - 4) * 32, lane);
      cHv0 = wmma32(a, load_b(WVq, 36, kb * 8, 0, lane), cHv0);
      cHv1 = wmma32(a, load_b(WVq, 36, kb * 8, 1, lane), cHv1);
    }
    // relu + stage h_a/h_v (ATT region now dead)
#pragma unroll
    for (int g = 0; g < 8; ++g) {
      const int M = g + hi * 8;
      stg[M * RSH + 288 + n16] = (half_t)fmaxf(cHa0[g], 0.f);
      stg[M * RSH + 304 + n16] = (half_t)fmaxf(cHa1[g], 0.f);
      stg[M * RSH + 320 + n16] = (half_t)fmaxf(cHv0[g], 0.f);
      stg[M * RSH + 336 + n16] = (half_t)fmaxf(cHv1[g], 0.f);
    }
    lds_fence();

    // ---- att_a = [h_a,aud] @ [w_ha;e3]^T + b ; att_v likewise --------------
    v8f ca = zero, cv = zero;
#pragma unroll
    for (int kb = 0; kb < 5; ++kb) {
      const int coA = (kb == 0) ? 288 : (kb - 1) * 32;
      const int coV = (kb == 0) ? 320 : 128 + (kb - 1) * 32;
      ca = wmma32(load_a(stg, coA, lane), load_b(HAq, 20, kb * 8, 0, lane), ca);
      cv = wmma32(load_a(stg, coV, lane), load_b(HVq, 20, kb * 8, 0, lane), cv);
    }
    {
      const float ba = fl[OFF_E3B + n16], bv = fl[OFF_E4B + n16];
#pragma unroll
      for (int g = 0; g < 8; ++g) {
        const int M = g + hi * 8;
        if (n16 < 8) {
          stg[M * RSH + n16] = (half_t)(ca[g] + ba);       // avf[0..7]
          stg[M * RSH + 8 + n16] = (half_t)(cv[g] + bv);   // avf[8..15]
        }
        stg[M * RSH + 16 + n16] = (half_t)0.f;             // pad K 16..31
      }
    }
    lds_fence();

    // ---- hid = avf @ r1^T + b ; out = hid @ r2^T + b -----------------------
    v8f cD[8];
#pragma unroll
    for (int nb = 0; nb < 8; ++nb) cD[nb] = zero;
    {
      v16h a = load_a(stg, 0, lane);
#pragma unroll
      for (int nb = 0; nb < 8; ++nb)
        cD[nb] = wmma32(a, load_b(R1q, 132, 0, nb, lane), cD[nb]);
    }
    float part[8];
#pragma unroll
    for (int g = 0; g < 8; ++g) part[g] = 0.f;
#pragma unroll
    for (int nb = 0; nb < 8; ++nb) {
      const float w = fl[OFF_R2W + nb * 16 + n16];
      const float bb = fl[OFF_R1B + nb * 16 + n16];
#pragma unroll
      for (int g = 0; g < 8; ++g) part[g] = fmaf(cD[nb][g] + bb, w, part[g]);
    }
#pragma unroll
    for (int g = 0; g < 8; ++g) {
      float s = part[g];
      s += __shfl_xor(s, 1, 32); s += __shfl_xor(s, 2, 32);
      s += __shfl_xor(s, 4, 32); s += __shfl_xor(s, 8, 32);
      const int outrow = row0 + g + hi * 8;
      if (n16 == g && outrow < Btot) out[outrow] = s + r2bv;
    }

    // ---- prefetch next tile's inputs ---------------------------------------
    const int nt = tile + gridDim.x;
    if (nt < ntiles) {
      long prow = (long)nt * ROWS_PER_BLK + wave * 16 + r;
      if (prow > Btot - 1) prow = Btot - 1;
      __builtin_prefetch(f1 + (size_t)prow * 88);
      __builtin_prefetch(f2 + (size_t)prow * 168);
    }
  }
}

extern "C" void kernel_launch(void* const* d_in, const int* in_sizes, int n_in,
                              void* d_out, int out_size, void* d_ws, size_t ws_size,
                              hipStream_t stream) {
  (void)n_in; (void)out_size; (void)d_ws; (void)ws_size;
  const float* f1   = (const float*)d_in[0];
  const float* f2   = (const float*)d_in[1];
  const float* e1_w = (const float*)d_in[2];
  const float* e1_b = (const float*)d_in[3];
  const float* e2_w = (const float*)d_in[4];
  const float* e2_b = (const float*)d_in[5];
  const float* affa = (const float*)d_in[6];
  const float* affv = (const float*)d_in[7];
  const float* w_a  = (const float*)d_in[8];
  const float* w_v  = (const float*)d_in[9];
  const float* w_ca = (const float*)d_in[10];
  const float* w_cv = (const float*)d_in[11];
  const float* w_ha = (const float*)d_in[12];
  const float* w_hv = (const float*)d_in[13];
  const float* e3_w = (const float*)d_in[14];
  const float* e3_b = (const float*)d_in[15];
  const float* e4_w = (const float*)d_in[16];
  const float* e4_b = (const float*)d_in[17];
  const float* r1_w = (const float*)d_in[18];
  const float* r1_b = (const float*)d_in[19];
  const float* r2_w = (const float*)d_in[20];
  const float* r2_b = (const float*)d_in[21];

  const int Btot = in_sizes[0] / 88;
  const int ntiles = (Btot + ROWS_PER_BLK - 1) / ROWS_PER_BLK;
  int grid = ntiles < 1024 ? ntiles : 1024;

  (void)hipFuncSetAttribute(reinterpret_cast<const void*>(cam_fused),
                            hipFuncAttributeMaxDynamicSharedMemorySize, LDS_BYTES);

  cam_fused<<<grid, 128, LDS_BYTES, stream>>>(
      f1, f2, e1_w, e1_b, e2_w, e2_b, affa, affv, w_a, w_v, w_ca, w_cv,
      w_ha, w_hv, e3_w, e3_b, e4_w, e4_b, r1_w, r1_b, r2_w, r2_b,
      (float*)d_out, Btot, ntiles);
}
